// PointTransformer_50766513438986
// MI455X (gfx1250) — matrix operand order repaired and checked by reading, not compiled, not run on hardware
//
#include <hip/hip_runtime.h>
#include <math.h>

// ---------------------------------------------------------------------------
// CDNA5 WMMA types
// ---------------------------------------------------------------------------
typedef __attribute__((ext_vector_type(16))) __bf16 v16bf;
typedef __attribute__((ext_vector_type(8)))  float  v8f;
typedef __attribute__((ext_vector_type(4)))  float  f32x4;

__device__ __forceinline__ __bf16 f2bf(float f) {
  union { float f; unsigned u; } a; a.f = f;
  unsigned u = a.u + 0x7FFFu + ((a.u >> 16) & 1u);  // round-to-nearest-even
  union { unsigned short s; __bf16 b; } o; o.s = (unsigned short)(u >> 16);
  return o.b;
}

// ---------------------------------------------------------------------------
// GEMM: C[M,N] = act( A[M,K](lda) @ W[K,N] + bias + resid )
// Contract (enforced by the host pipeline):
//   * M % 16 == 0, N % 64 == 0             -> no M/N guards anywhere
//   * lda % 32 == 0, A rows zero-padded in [Kd, lda)  -> unconditional A loads
//   * B row index clamped to Kd-1; A's zero padding kills the bogus products
// One wave per 16x16 tile, 4 waves (4 N-tiles) per block.
// B tiles (32x16 f32 = 2KB) are DMA'd to LDS with GLOBAL_LOAD_ASYNC_TO_LDS_B128
// (ASYNCcnt), double buffered per wave: DMA of tile s+1 overlaps WMMA of tile s.
// Per-wave-private LDS regions -> no barriers, only s_wait_asynccnt.
// ---------------------------------------------------------------------------
__global__ void __launch_bounds__(128)
gemm_bf16_wmma(const float* __restrict__ A, int lda,
               const float* __restrict__ W,
               const float* __restrict__ bias, const float* __restrict__ resid,
               float* __restrict__ C, int M, int Kd, int Nd, int relu_act) {
  __shared__ float sB[4 * 2 * 512];          // 4 waves x 2 buffers x (32x16) f32
  const int lane = threadIdx.x & 31;
  const int wave = threadIdx.x >> 5;
  const int m0 = blockIdx.x * 16;
  const int n0 = (blockIdx.y * 4 + wave) * 16;
  const int mrow = lane & 15;
  const int hi   = lane >> 4;                // 0 or 1
  const int ncol = n0 + mrow;                // C store column
  const float* Arow = A + (size_t)(m0 + mrow) * lda;
  float* sBw = &sB[wave * 1024];             // this wave's two 512-float buffers
  const int ksteps = (Kd + 31) >> 5;

  // DMA one 32x16 B tile (k rows kb..kb+31, cols n0..n0+15) into LDS buffer.
  // Chunk u = i*32+lane copies 16 bytes: row u>>2 (clamped), quarter u&3.
  auto issueB = [&](int s, int buf) {
    const int kb = s * 32;
#pragma unroll
    for (int i = 0; i < 4; ++i) {
      int u = i * 32 + lane;
      int krow = kb + (u >> 2);
      krow = (krow < Kd) ? krow : (Kd - 1);
      const float* gp = W + (size_t)krow * Nd + n0 + (u & 3) * 4;
      unsigned ldsoff = (unsigned)(size_t)&sBw[buf * 512 + u * 4];
      asm volatile("global_load_async_to_lds_b128 %0, %1, off"
                   :: "v"(ldsoff), "v"((unsigned long long)(size_t)gp)
                   : "memory");
    }
  };

  v8f acc = {0.f, 0.f, 0.f, 0.f, 0.f, 0.f, 0.f, 0.f};
  issueB(0, 0);

  for (int s = 0; s < ksteps; ++s) {
    const int kb = s * 32;
    const int kA0 = kb + hi * 8;
    // A: elems 0..7 -> K = kA0..kA0+7 ; elems 8..15 -> K = kA0+16..kA0+23
    f32x4 a0 = *(const f32x4*)(Arow + kA0);
    f32x4 a1 = *(const f32x4*)(Arow + kA0 + 4);
    f32x4 a2 = *(const f32x4*)(Arow + kA0 + 16);
    f32x4 a3 = *(const f32x4*)(Arow + kA0 + 20);
    __builtin_prefetch(Arow + kb + 32, 0, 0);
    v16bf av, bv;
#pragma unroll
    for (int e = 0; e < 4; ++e) {
      av[e]      = f2bf(a0[e]);
      av[e + 4]  = f2bf(a1[e]);
      av[e + 8]  = f2bf(a2[e]);
      av[e + 12] = f2bf(a3[e]);
    }
    // kick off next tile's DMA, then wait for this tile's 4 copies
    if (s + 1 < ksteps) {
      issueB(s + 1, (s + 1) & 1);
      asm volatile("s_wait_asynccnt 0x4" ::: "memory");
    } else {
      asm volatile("s_wait_asynccnt 0x0" ::: "memory");
    }
    // B from LDS: elems 0..7 -> local k = hi*8+e ; elems 8..15 -> k = hi*8+e+8, column = lane&15
    const float* sT = &sBw[(s & 1) * 512];
    const int kB0l = hi * 8;
#pragma unroll
    for (int e = 0; e < 16; ++e)
      bv[e] = f2bf(sT[(kB0l + e + (e >= 8 ? 8 : 0)) * 16 + (lane & 15)]);
    acc = __builtin_amdgcn_wmma_f32_16x16x32_bf16(
        /*neg_a=*/false, av, /*neg_b=*/false, bv,
        /*c_mod=*/(short)0, acc, /*reuse_a=*/false, /*reuse_b=*/false);
  }

  // Epilogue: wave-uniform branches hoisted out of the per-element loop.
  const int mb = m0 + hi * 8;
  float badd = bias ? bias[ncol] : 0.f;
  float vr[8];
#pragma unroll
  for (int r = 0; r < 8; ++r) vr[r] = acc[r] + badd;
  if (resid) {
    const float* Rp = resid + (size_t)mb * Nd + ncol;
#pragma unroll
    for (int r = 0; r < 8; ++r) vr[r] += Rp[(size_t)r * Nd];
  }
  if (relu_act) {
#pragma unroll
    for (int r = 0; r < 8; ++r) vr[r] = fmaxf(vr[r], 0.f);
  }
  float* Cp = C + (size_t)mb * Nd + ncol;
#pragma unroll
  for (int r = 0; r < 8; ++r) Cp[(size_t)r * Nd] = vr[r];
}

// ---------------------------------------------------------------------------
// Column-wise BatchNorm (training, identity affine), strided in/out.
// Launch with gridDim.x == ldy; blocks with c >= d just zero the pad column.
// ---------------------------------------------------------------------------
__global__ void bn_col(const float* __restrict__ x, int ldx,
                       float* __restrict__ y, int ldy, int n, int d) {
  __shared__ float s1[256], s2[256];
  int c = blockIdx.x, t = threadIdx.x;
  if (c >= d) {                                  // zero pad column
    for (int i = t; i < n; i += 256) y[(size_t)i * ldy + c] = 0.f;
    return;
  }
  float a = 0.f, b = 0.f;
  for (int i = t; i < n; i += 256) { float v = x[(size_t)i * ldx + c]; a += v; b += v * v; }
  s1[t] = a; s2[t] = b; __syncthreads();
  for (int o = 128; o > 0; o >>= 1) {
    if (t < o) { s1[t] += s1[t + o]; s2[t] += s2[t + o]; }
    __syncthreads();
  }
  float mean = s1[0] / (float)n;
  float var  = s2[0] / (float)n - mean * mean;
  float inv  = rsqrtf(var + 1e-5f);
  for (int i = t; i < n; i += 256)
    y[(size_t)i * ldy + c] = (x[(size_t)i * ldx + c] - mean) * inv;
}

// ---------------------------------------------------------------------------
// Brute-force kNN (K=7) with per-thread register top-k; excl==1 skips self.
// ---------------------------------------------------------------------------
__global__ void knn_brute(const float* __restrict__ q, int nq,
                          const float* __restrict__ r, int nr,
                          int* __restrict__ nbr, int excl) {
  int i = blockIdx.x * blockDim.x + threadIdx.x;
  if (i >= nq) return;
  float qx = q[3 * i], qy = q[3 * i + 1], qz = q[3 * i + 2];
  float bd[7]; int bi[7];
#pragma unroll
  for (int j = 0; j < 7; ++j) { bd[j] = 1e30f; bi[j] = 0; }
  for (int j = 0; j < nr; ++j) {
    if (excl && j == i) continue;
    float dx = r[3 * j] - qx, dy = r[3 * j + 1] - qy, dz = r[3 * j + 2] - qz;
    float d = dx * dx + dy * dy + dz * dz;
    if (d < bd[6]) {
      int p = 6;
      while (p > 0 && bd[p - 1] > d) { bd[p] = bd[p - 1]; bi[p] = bi[p - 1]; --p; }
      bd[p] = d; bi[p] = j;
    }
  }
#pragma unroll
  for (int j = 0; j < 7; ++j) nbr[i * 7 + j] = bi[j];
}

// ---------------------------------------------------------------------------
// Farthest point sampling: single persistent block, sequential argmax scan.
// up[0]=0 (reference scan emits `last` before computing the new argmax).
// Then stable compaction of non-selected indices into `down`.
// ---------------------------------------------------------------------------
__global__ void __launch_bounds__(1024)
fps_seq(const float* __restrict__ pos, int n, int n_up,
        int* __restrict__ up, int* __restrict__ down,
        float* __restrict__ d2b, int* __restrict__ flags) {
  __shared__ float sv[1024];
  __shared__ int   si[1024];
  int t = threadIdx.x;
  for (int i = t; i < n; i += 1024) { d2b[i] = 1e10f; flags[i] = 0; }
  __syncthreads();
  int last = 0;
  for (int s = 0; s < n_up; ++s) {
    if (t == 0) { up[s] = last; flags[last] = 1; }
    __syncthreads();
    float lx = pos[3 * last], ly = pos[3 * last + 1], lz = pos[3 * last + 2];
    float best = -1.f; int bidx = n;
    for (int i = t; i < n; i += 1024) {
      float dx = pos[3 * i] - lx, dy = pos[3 * i + 1] - ly, dz = pos[3 * i + 2] - lz;
      float dd = fminf(d2b[i], dx * dx + dy * dy + dz * dz);
      d2b[i] = dd;
      if (dd > best) { best = dd; bidx = i; }   // strict > keeps first max
    }
    sv[t] = best; si[t] = bidx; __syncthreads();
    for (int o = 512; o > 0; o >>= 1) {
      if (t < o) {
        if (sv[t + o] > sv[t] || (sv[t + o] == sv[t] && si[t + o] < si[t])) {
          sv[t] = sv[t + o]; si[t] = si[t + o];
        }
      }
      __syncthreads();
    }
    last = si[0];
    __syncthreads();
  }
  if (t == 0) {
    int c = 0;
    for (int i = 0; i < n; ++i) if (!flags[i]) down[c++] = i;
  }
}

// ---------------------------------------------------------------------------
// Graph-transformer attention over the 7-neighbor graph (1 head):
// s_j = exp(clip(<K[src_j],Q[i]>/sqrt(d),-5,5)); a[i] = sum s_j V[src_j]/(sum+1e-6)
// ---------------------------------------------------------------------------
__global__ void gt_attn(const float* __restrict__ Q, const float* __restrict__ Kf,
                        const float* __restrict__ V, const int* __restrict__ nbr,
                        float* __restrict__ out, int n, int d) {
  __shared__ float red[256];
  __shared__ float sw[8];
  int i = blockIdx.x, t = threadIdx.x;
  float scale = rsqrtf((float)d);
  for (int j = 0; j < 7; ++j) {
    int s = nbr[i * 7 + j];
    float p = 0.f;
    for (int c = t; c < d; c += 256) p += Kf[(size_t)s * d + c] * Q[(size_t)i * d + c];
    red[t] = p; __syncthreads();
    for (int o = 128; o > 0; o >>= 1) { if (t < o) red[t] += red[t + o]; __syncthreads(); }
    if (t == 0) {
      float sc = red[0] * scale;
      sc = fminf(5.f, fmaxf(-5.f, sc));
      sw[j] = expf(sc);
    }
    __syncthreads();
  }
  float den = 1e-6f;
#pragma unroll
  for (int j = 0; j < 7; ++j) den += sw[j];
  for (int c = t; c < d; c += 256) {
    float num = 0.f;
#pragma unroll
    for (int j = 0; j < 7; ++j) num += sw[j] * V[(size_t)nbr[i * 7 + j] * d + c];
    out[(size_t)i * d + c] = num / den;
  }
}

// ---------------------------------------------------------------------------
// Transition-down mailbox with padded row stride dinp (pads zero-filled):
// row (i,j) = [ pos[down[nb]] - pos[up[i]] || h[down[nb]] || 0-pad ]
// ---------------------------------------------------------------------------
__global__ void mailbox(const float* __restrict__ pos, const float* __restrict__ h,
                        const int* __restrict__ up, const int* __restrict__ down,
                        const int* __restrict__ nbr, float* __restrict__ M,
                        int d, int dinp) {
  int rrow = blockIdx.x;               // n_up*7 rows
  int i = rrow / 7;
  int dn = down[nbr[rrow]];
  int u  = up[i];
  float* Mr = M + (size_t)rrow * dinp;
  if (threadIdx.x < 3)
    Mr[threadIdx.x] = pos[3 * dn + threadIdx.x] - pos[3 * u + threadIdx.x];
  for (int c = threadIdx.x; c < dinp - 3; c += blockDim.x)
    Mr[3 + c] = (c < d) ? h[(size_t)dn * d + c] : 0.f;
}

__global__ void maxpool7(const float* __restrict__ t, float* __restrict__ o,
                         int n_up, int dout) {
  int idx = blockIdx.x * blockDim.x + threadIdx.x;
  if (idx >= n_up * dout) return;
  int i = idx / dout, c = idx % dout;
  const float* b = t + ((size_t)i * 7) * dout + c;
  float m = b[0];
#pragma unroll
  for (int j = 1; j < 7; ++j) m = fmaxf(m, b[(size_t)j * dout]);
  o[idx] = m;
}

__global__ void gather_rows(const float* __restrict__ src, const int* __restrict__ idx,
                            float* __restrict__ dst, int nrows, int ncols) {
  int t = blockIdx.x * blockDim.x + threadIdx.x;
  if (t >= nrows * ncols) return;
  int r = t / ncols, c = t % ncols;
  dst[t] = src[(size_t)idx[r] * ncols + c];
}

// ---------------------------------------------------------------------------
// Host orchestration
// ---------------------------------------------------------------------------
struct GTParams { const float *Wq, *Wk, *Wv, *Wo, *Wf1, *bf1, *Wf2, *bf2; };
struct BlkParams { const float *mW1, *mb1, *mW2, *mb2; GTParams tf; };

extern "C" void kernel_launch(void* const* d_in, const int* in_sizes, int n_in,
                              void* d_out, int out_size, void* d_ws, size_t ws_size,
                              hipStream_t stream) {
  (void)n_in; (void)out_size; (void)ws_size;
  auto F = [&](int i) { return (const float*)d_in[i]; };

  const int n0 = in_sizes[0] / 8;            // 8192
  const float* h_in   = F(0);
  const float* pos_in = F(1);
  // params in setup_inputs() insertion order:
  const float *fcW1 = F(2), *fcb1 = F(3), *fcW2 = F(4), *fcb2 = F(5);
  GTParams ptb{F(6), F(7), F(8), F(9), F(10), F(11), F(12), F(13)};
  BlkParams blks[4];
  int pi = 14;
  for (int b = 0; b < 4; ++b) {
    blks[b].mW1 = F(pi++); blks[b].mb1 = F(pi++);
    blks[b].mW2 = F(pi++); blks[b].mb2 = F(pi++);
    blks[b].tf = GTParams{F(pi++), F(pi++), F(pi++), F(pi++),
                          F(pi++), F(pi++), F(pi++), F(pi++)};
  }

  // ---- workspace arena (bump allocator, 256B aligned blocks) ----
  size_t off = 0;
  auto allocf = [&](size_t elems) -> float* {
    off = (off + 255) & ~(size_t)255;
    float* p = (float*)((char*)d_ws + off); off += elems * sizeof(float); return p;
  };
  auto alloci = [&](size_t elems) -> int* {
    off = (off + 255) & ~(size_t)255;
    int* p = (int*)((char*)d_ws + off); off += elems * sizeof(int); return p;
  };

  float* hbuf = allocf((size_t)n0 * 64);            // current hidden state
  float* t1   = allocf((size_t)n0 * 64);
  float* t2   = allocf((size_t)n0 * 64);
  float* qb   = allocf((size_t)n0 * 64);
  float* kb   = allocf((size_t)n0 * 64);
  float* vb   = allocf((size_t)n0 * 64);
  float* ab   = allocf((size_t)n0 * 64);
  float* fb   = allocf((size_t)n0 * 128);           // FFN hidden [n,2d]
  float* mb   = allocf((size_t)(n0 / 4) * 7 * 96);  // mailbox, padded stride
  float* mt1  = allocf((size_t)(n0 / 4) * 7 * 128);
  float* mt2  = allocf((size_t)(n0 / 4) * 7 * 128);
  float* posA = allocf((size_t)(n0 / 4) * 3);
  float* posB = allocf((size_t)(n0 / 4) * 3);
  float* posD = allocf((size_t)n0 * 3);
  float* d2b  = allocf((size_t)n0);
  int*   nbr   = alloci((size_t)n0 * 7);
  int*   up    = alloci((size_t)n0 / 4);
  int*   down  = alloci((size_t)n0);
  int*   flags = alloci((size_t)n0);

  auto gemm = [&](const float* A, int lda, const float* W, const float* bias,
                  const float* resid, float* C, int M, int Kd, int Nd, int relu) {
    dim3 g((unsigned)((M + 15) / 16), (unsigned)((Nd + 63) / 64));
    gemm_bf16_wmma<<<g, dim3(128), 0, stream>>>(A, lda, W, bias, resid, C,
                                                M, Kd, Nd, relu);
  };

  auto gtlayer = [&](float* h, int nn, int dd, const GTParams& p) {
    gemm(h, dd, p.Wq, nullptr, nullptr, qb, nn, dd, dd, 0);
    gemm(h, dd, p.Wk, nullptr, nullptr, kb, nn, dd, dd, 0);
    gemm(h, dd, p.Wv, nullptr, nullptr, vb, nn, dd, dd, 0);
    gt_attn<<<nn, 256, 0, stream>>>(qb, kb, vb, nbr, ab, nn, dd);
    gemm(ab, dd, p.Wo, nullptr, /*resid=*/h, t1, nn, dd, dd, 0);   // h + a@Wo
    bn_col<<<dd, 256, 0, stream>>>(t1, dd, t1, dd, nn, dd);        // h1
    gemm(t1, dd, p.Wf1, p.bf1, nullptr, fb, nn, dd, 2 * dd, 1);    // relu(h1@Wf1+b)
    gemm(fb, 2 * dd, p.Wf2, p.bf2, /*resid=*/t1, t2, nn, 2 * dd, dd, 0); // h1 + f
    bn_col<<<dd, 256, 0, stream>>>(t2, dd, h, dd, nn, dd);
  };

  // ---- stage 0: bn -> fc -> knn graph -> transformer ----
  // BN of the 8-wide input goes to a 32-stride zero-padded buffer so the
  // fc1 GEMM (Kd=8) needs no K guards.
  bn_col<<<32, 256, 0, stream>>>(h_in, 8, t1, 32, n0, 8);
  gemm(t1, 32, fcW1, fcb1, nullptr, t2, n0, 8, 64, 1);
  gemm(t2, 64, fcW2, fcb2, nullptr, hbuf, n0, 64, 64, 0);
  knn_brute<<<(n0 + 127) / 128, 128, 0, stream>>>(pos_in, n0, pos_in, n0, nbr, 1);
  gtlayer(hbuf, n0, 64, ptb);

  // ---- 4 transition-down + transformer blocks ----
  const float* pos = pos_in;
  int n = n0, d = 64;
  for (int b = 0; b < 4; ++b) {
    int n_up = n / 4, dout = d * 2, nd = n - n_up;
    int din  = d + 3;
    int dinp = (din + 31) & ~31;                 // padded mailbox stride
    fps_seq<<<1, 1024, 0, stream>>>(pos, n, n_up, up, down, d2b, flags);
    float* posNew = (b & 1) ? posB : posA;
    gather_rows<<<(n_up * 3 + 255) / 256, 256, 0, stream>>>(pos, up, posNew, n_up, 3);
    gather_rows<<<(nd * 3 + 255) / 256, 256, 0, stream>>>(pos, down, posD, nd, 3);
    knn_brute<<<(n_up + 127) / 128, 128, 0, stream>>>(posNew, n_up, posD, nd, nbr, 0);
    mailbox<<<n_up * 7, 128, 0, stream>>>(pos, hbuf, up, down, nbr, mb, d, dinp);
    gemm(mb, dinp, blks[b].mW1, blks[b].mb1, nullptr, mt1, n_up * 7, din, dout, 1);
    gemm(mt1, dout, blks[b].mW2, blks[b].mb2, nullptr, mt2, n_up * 7, dout, dout, 1);
    maxpool7<<<(n_up * dout + 255) / 256, 256, 0, stream>>>(mt2, hbuf, n_up, dout);
    pos = posNew; n = n_up; d = dout;
    knn_brute<<<(n + 127) / 128, 128, 0, stream>>>(pos, n, pos, n, nbr, 1);
    gtlayer(hbuf, n, d, blks[b].tf);
  }

  // final h: [32, 1024]
  hipMemcpyAsync(d_out, hbuf, (size_t)n * d * sizeof(float),
                 hipMemcpyDeviceToDevice, stream);
}
